// CrossAttention_16303695855849
// MI455X (gfx1250) — matrix operand order repaired and checked by reading, not compiled, not run on hardware
//
#include <hip/hip_runtime.h>

// ---------------------------------------------------------------------------
// CDNA5 (gfx1250, wave32) cross-attention pipeline:
//   v_wmma_f32_16x16x32_f16 matrix ops + async global->LDS staging (ASYNCcnt)
// ---------------------------------------------------------------------------

typedef _Float16 half_t;
typedef __attribute__((ext_vector_type(16))) _Float16 v16h;
typedef __attribute__((ext_vector_type(8)))  _Float16 v8h;
typedef __attribute__((ext_vector_type(8)))  float    v8f;

__device__ __forceinline__ v16h cat8(v8h lo, v8h hi) {
  return __builtin_shufflevector(lo, hi, 0,1,2,3,4,5,6,7,8,9,10,11,12,13,14,15);
}

__device__ __forceinline__ v8f vzero8() {
  v8f z;
#pragma unroll
  for (int i = 0; i < 8; ++i) z[i] = 0.0f;
  return z;
}

__device__ __forceinline__ v8f wmma_f16(v16h a, v16h b, v8f c) {
  // D = A(16x32 f16) * B(32x16 f16) + C(16x16 f32)
  return __builtin_amdgcn_wmma_f32_16x16x32_f16(false, a, false, b, (short)0, c,
                                                false, false);
}

// A-fragment (16x32, M x K) from row-major [M][K] storage (stride in halves).
__device__ __forceinline__ v16h load_frag_a(const half_t* p, int stride, int row,
                                            int kb, int lane) {
  const half_t* q = p + (size_t)row * stride + kb + ((lane >> 4) << 3);
  v8h lo = *(const v8h*)(q);
  v8h hi = *(const v8h*)(q + 16);
  return cat8(lo, hi);
}

// B-fragment (32x16, K x N) from row-major [N][K] (K-contiguous) storage.
__device__ __forceinline__ v16h load_frag_b(const half_t* p, int stride, int col,
                                            int kb, int lane) {
  const half_t* q = p + (size_t)col * stride + kb + ((lane >> 4) << 3);
  v8h lo = *(const v8h*)(q);
  v8h hi = *(const v8h*)(q + 16);
  return cat8(lo, hi);
}

// ---- CDNA5 async global->LDS DMA (tracked by ASYNCcnt; no VGPR staging) ----
// Generic pointers to LDS carry the LDS byte offset in their low 32 bits
// (aperture truncation rule), which is exactly what VDST wants.
__device__ __forceinline__ void async_copy_b128(void* lds_dst, const void* gsrc) {
  unsigned l = (unsigned)(unsigned long long)(uintptr_t)lds_dst;
  asm volatile("global_load_async_to_lds_b128 %0, %1, off"
               :: "v"(l), "v"(gsrc) : "memory");
}

__device__ __forceinline__ void wait_async0() {
  asm volatile("s_wait_asynccnt 0x0" ::: "memory");
}

// ---------------------------------------------------------------------------
// Kernel 0: convert weights to f16 (kv conv weights reordered [oc][t][ic])
// ---------------------------------------------------------------------------
__global__ __launch_bounds__(256) void convert_weights_kernel(
    const float* __restrict__ q_w, const float* __restrict__ kv_w,
    const float* __restrict__ out_w, half_t* __restrict__ qw_h,
    half_t* __restrict__ kvw_h, half_t* __restrict__ outw_h) {
  int i = blockIdx.x * 256 + threadIdx.x;
  if (i < 1024 * 128) qw_h[i] = (half_t)q_w[i];
  if (i < 128 * 1024) outw_h[i] = (half_t)out_w[i];
  if (i < 2048 * 768) {
    int oc = i / 768, rem = i - oc * 768;
    int tt = rem >> 8, ic = rem & 255;
    kvw_h[i] = (half_t)kv_w[(oc * 256 + ic) * 3 + tt];
  }
}

// ---------------------------------------------------------------------------
// GroupNorm stats: one block per (batch, group); group data is contiguous.
// stats[bg*2] = mean, stats[bg*2+1] = rsqrt(var + eps)
// ---------------------------------------------------------------------------
__global__ __launch_bounds__(256) void gn_stats_kernel(
    const float* __restrict__ src, float* __restrict__ stats, int C, int chpg) {
  const int L = 2048;
  int bg = blockIdx.x;
  int b = bg >> 5, g = bg & 31;
  const float* p = src + ((size_t)b * C + (size_t)g * chpg) * L;
  int n = chpg * L;
  float s = 0.f, ss = 0.f;
  for (int i = threadIdx.x; i < n; i += 256) {
    float v = p[i];
    s += v;
    ss += v * v;
  }
#pragma unroll
  for (int o = 16; o > 0; o >>= 1) {
    s += __shfl_xor(s, o, 32);
    ss += __shfl_xor(ss, o, 32);
  }
  __shared__ float rs[8], rss[8];
  if ((threadIdx.x & 31) == 0) {
    rs[threadIdx.x >> 5] = s;
    rss[threadIdx.x >> 5] = ss;
  }
  __syncthreads();
  if (threadIdx.x == 0) {
    float S = 0.f, SS = 0.f;
#pragma unroll
    for (int i = 0; i < 8; ++i) { S += rs[i]; SS += rss[i]; }
    float mu = S / (float)n;
    float var = SS / (float)n - mu * mu;
    stats[bg * 2] = mu;
    stats[bg * 2 + 1] = rsqrtf(var + 1e-5f);
  }
}

// ---------------------------------------------------------------------------
// Kernel: q projection. GN1 + Swish fused into LDS staging, then WMMA GEMM.
// Stores q * (1/sqrt(128)) transposed as q_ws[bh][lq][c] (f16).
// ---------------------------------------------------------------------------
__global__ __launch_bounds__(256) void qproj_kernel(
    const float* __restrict__ x, const float* __restrict__ gn1_w,
    const float* __restrict__ gn1_b, const float* __restrict__ stats1,
    const half_t* __restrict__ qw_h, const float* __restrict__ q_b,
    half_t* __restrict__ q_ws) {
  __shared__ alignas(16) half_t Blds[128][136];  // [lq_local][c]
  int b = blockIdx.z, Mt = blockIdx.y, Nt = blockIdx.x;
  int Mbase = Mt * 128, Nbase = Nt * 128;
  int t = threadIdx.x, wave = t >> 5, lane = t & 31;
  int mw = wave >> 2, nw = wave & 3;

  const float* xb = x + (size_t)b * 128 * 2048 + Nbase;
  for (int i = t; i < 128 * 128; i += 256) {
    int c = i >> 7, j = i & 127;
    float mu = stats1[(b * 32 + (c >> 2)) * 2];
    float rstd = stats1[(b * 32 + (c >> 2)) * 2 + 1];
    float vx = xb[(size_t)c * 2048 + j];
    float hn = (vx - mu) * rstd * gn1_w[c] + gn1_b[c];
    float hs = hn / (1.f + __expf(-hn));  // swish
    Blds[j][c] = (half_t)hs;
  }
  __syncthreads();

  v8f acc[4][2];
#pragma unroll
  for (int mt = 0; mt < 4; ++mt)
#pragma unroll
    for (int nt = 0; nt < 2; ++nt) acc[mt][nt] = vzero8();

#pragma unroll
  for (int cb = 0; cb < 128; cb += 32) {
    v16h bf[2];
#pragma unroll
    for (int nt = 0; nt < 2; ++nt)
      bf[nt] = load_frag_b(&Blds[0][0], 136, nw * 32 + nt * 16 + (lane & 15), cb, lane);
#pragma unroll
    for (int mt = 0; mt < 4; ++mt) {
      int o = Mbase + mw * 64 + mt * 16 + (lane & 15);
      v16h af = load_frag_a(qw_h, 128, o, cb, lane);
#pragma unroll
      for (int nt = 0; nt < 2; ++nt) acc[mt][nt] = wmma_f16(af, bf[nt], acc[mt][nt]);
    }
  }

  const float scale = 0.08838834764831845f;  // 1/sqrt(128), folded into q
#pragma unroll
  for (int mt = 0; mt < 4; ++mt)
#pragma unroll
    for (int nt = 0; nt < 2; ++nt) {
      int o0 = Mbase + mw * 64 + mt * 16 + ((lane >> 4) << 3);
      int lq = Nbase + nw * 32 + nt * 16 + (lane & 15);
      int bh = b * 8 + (o0 >> 7), c0 = o0 & 127;
      v8h st;
#pragma unroll
      for (int v = 0; v < 8; ++v)
        st[v] = (half_t)((acc[mt][nt][v] + q_b[o0 + v]) * scale);
      *(v8h*)(q_ws + ((size_t)bh * 2048 + lq) * 128 + c0) = st;
    }
}

// ---------------------------------------------------------------------------
// Kernel: kv projection, Conv1d(k=3,pad=1) as implicit-im2col WMMA GEMM.
// rows 0..1023 -> K (stored transposed [bh][lk][c]); 1024..2047 -> V [bh][c][lk].
// ---------------------------------------------------------------------------
__global__ __launch_bounds__(256) void kvproj_kernel(
    const float* __restrict__ ctx, const half_t* __restrict__ kvw_h,
    const float* __restrict__ kv_b, half_t* __restrict__ k_ws,
    half_t* __restrict__ v_ws) {
  __shared__ alignas(16) half_t Clds[130][264];  // [l_halo][ic]
  int b = blockIdx.z, Mt = blockIdx.y, Nt = blockIdx.x;
  int Mbase = Mt * 128, Nbase = Nt * 128;
  int t = threadIdx.x, wave = t >> 5, lane = t & 31;
  int mw = wave >> 2, nw = wave & 3;

  const float* cg = ctx + (size_t)b * 256 * 2048;
  for (int i = t; i < 256 * 130; i += 256) {
    int ic = i / 130, r = i - ic * 130;
    int l = Nbase - 1 + r;
    float vv = (l >= 0 && l < 2048) ? cg[(size_t)ic * 2048 + l] : 0.f;
    Clds[r][ic] = (half_t)vv;
  }
  __syncthreads();

  v8f acc[4][2];
#pragma unroll
  for (int mt = 0; mt < 4; ++mt)
#pragma unroll
    for (int nt = 0; nt < 2; ++nt) acc[mt][nt] = vzero8();

  for (int s = 0; s < 24; ++s) {
    int tc = s >> 3, icb = (s & 7) * 32;
    v16h bf[2];
#pragma unroll
    for (int nt = 0; nt < 2; ++nt) {
      int r = nw * 32 + nt * 16 + (lane & 15) + tc;  // l = l0 + n + tc - 1
      bf[nt] = load_frag_b(&Clds[0][0], 264, r, icb, lane);
    }
#pragma unroll
    for (int mt = 0; mt < 4; ++mt) {
      int oc = Mbase + mw * 64 + mt * 16 + (lane & 15);
      v16h af = load_frag_a(kvw_h, 768, oc, tc * 256 + icb, lane);
#pragma unroll
      for (int nt = 0; nt < 2; ++nt) acc[mt][nt] = wmma_f16(af, bf[nt], acc[mt][nt]);
    }
  }

#pragma unroll
  for (int mt = 0; mt < 4; ++mt)
#pragma unroll
    for (int nt = 0; nt < 2; ++nt) {
      int oc0 = Mbase + mw * 64 + mt * 16 + ((lane >> 4) << 3);
      int lk = Nbase + nw * 32 + nt * 16 + (lane & 15);
      if (oc0 < 1024) {  // K half -> transposed [bh][lk][c]
        int bh = b * 8 + (oc0 >> 7), c0 = oc0 & 127;
        v8h st;
#pragma unroll
        for (int v = 0; v < 8; ++v) st[v] = (half_t)(acc[mt][nt][v] + kv_b[oc0 + v]);
        *(v8h*)(k_ws + ((size_t)bh * 2048 + lk) * 128 + c0) = st;
      } else {           // V half -> natural [bh][c][lk]
        int oc1 = oc0 - 1024;
        int bh = b * 8 + (oc1 >> 7), c0 = oc1 & 127;
#pragma unroll
        for (int v = 0; v < 8; ++v)
          v_ws[((size_t)bh * 128 + c0 + v) * 2048 + lk] =
              (half_t)(acc[mt][nt][v] + kv_b[oc0 + v]);
      }
    }
}

// ---------------------------------------------------------------------------
// Kernel: flash attention with double-buffered ASYNC global->LDS K/V staging.
// Block = (bh, 128-query tile); 8 waves, wave owns 16 queries.
// Schedule per 64-lk chunk:
//   s_wait_asynccnt 0 (own DMAs done) -> barrier (publish all waves' DMAs)
//   -> issue next chunk into the other buffer -> 32 WMMAs on current buffer.
// ---------------------------------------------------------------------------
__global__ __launch_bounds__(256) void attn_kernel(
    const half_t* __restrict__ q_ws, const half_t* __restrict__ k_ws,
    const half_t* __restrict__ v_ws, float* __restrict__ o_ws) {
  __shared__ alignas(16) half_t Klds[2][64][136];    // [buf][lk][c]  (K^T rows)
  __shared__ alignas(16) half_t Vlds[2][128][72];    // [buf][c][lk]
  __shared__ alignas(16) half_t Plds[8][16][72];     // per-wave [lq][lk]

  int bh = blockIdx.x, qt = blockIdx.y;
  int b = bh >> 3, hh = bh & 7;
  int t = threadIdx.x, wave = t >> 5, lane = t & 31;
  int q0 = qt * 128;
  int lqn = q0 + wave * 16 + (lane & 15);

  const half_t* kgb = k_ws + (size_t)bh * 2048 * 128;
  const half_t* vgb = v_ws + (size_t)bh * 128 * 2048;

  // Issue async DMA for one 64-wide lk chunk into buffer `buf`.
  auto stage = [&](int buf, int lk0) {
#pragma unroll
    for (int j = 0; j < 4; ++j) {
      int i = t + 256 * j;               // 1024 b128 segments of K chunk
      int row = i >> 4, off = (i & 15) << 3;
      async_copy_b128(&Klds[buf][row][off],
                      kgb + ((size_t)(lk0 + row)) * 128 + off);
    }
#pragma unroll
    for (int j = 0; j < 4; ++j) {
      int i = t + 256 * j;               // 1024 b128 segments of V chunk
      int c = i >> 3, off = (i & 7) << 3;
      async_copy_b128(&Vlds[buf][c][off],
                      vgb + (size_t)c * 2048 + lk0 + off);
    }
  };

  // Preload Q fragments for the 4 c-steps (softmax scale already folded in)
  const half_t* qbase = q_ws + ((size_t)bh * 2048 + q0) * 128;
  v16h qf[4];
#pragma unroll
  for (int s = 0; s < 4; ++s)
    qf[s] = load_frag_b(qbase, 128, wave * 16 + (lane & 15), s * 32, lane);

  v8f oacc[8];
#pragma unroll
  for (int i = 0; i < 8; ++i) oacc[i] = vzero8();
  float mrun = -1e30f, lrun = 0.f;

  stage(0, 0);  // prologue DMA
  int buf = 0;

  for (int lk0 = 0; lk0 < 2048; lk0 += 64) {
    wait_async0();        // this wave's DMAs for `buf` complete
    __syncthreads();      // publish every wave's DMAs; buf^1 readers done
    if (lk0 + 64 < 2048) stage(buf ^ 1, lk0 + 64);

    // ---- S = K^T Q for this wave's 16 queries (4 lk-tiles x 4 c-steps) ----
    v8f sa[4];
#pragma unroll
    for (int mt = 0; mt < 4; ++mt) {
      v8f s = vzero8();
#pragma unroll
      for (int cs = 0; cs < 4; ++cs) {
        v16h a = load_frag_a(&Klds[buf][0][0], 136, mt * 16 + (lane & 15),
                             cs * 32, lane);
        s = wmma_f16(a, qf[cs], s);
      }
      sa[mt] = s;
    }

    // ---- online softmax over the 64-lk chunk ----
    float mc = -1e30f;
#pragma unroll
    for (int mt = 0; mt < 4; ++mt)
#pragma unroll
      for (int v = 0; v < 8; ++v) mc = fmaxf(mc, sa[mt][v]);
    mc = fmaxf(mc, __shfl_xor(mc, 16, 32));
    float mnew = fmaxf(mrun, mc);
    float alpha = __expf(mrun - mnew);
#pragma unroll
    for (int i = 0; i < 8; ++i)
#pragma unroll
      for (int v = 0; v < 8; ++v) oacc[i][v] *= alpha;

    float ls = 0.f;
#pragma unroll
    for (int mt = 0; mt < 4; ++mt) {
      v8h ph;
#pragma unroll
      for (int v = 0; v < 8; ++v) {
        float p = __expf(sa[mt][v] - mnew);
        ls += p;
        ph[v] = (half_t)p;
      }
      // P[lk][n] -> Plds[wave][n][lk], 8 contiguous lk per lane-half
      *(v8h*)&Plds[wave][lane & 15][mt * 16 + ((lane >> 4) << 3)] = ph;
    }
    ls += __shfl_xor(ls, 16, 32);
    lrun = lrun * alpha + ls;
    mrun = mnew;

    // ---- O += V * P (8 c-tiles x 2 lk-steps of 32) ----
#pragma unroll
    for (int ks = 0; ks < 2; ++ks) {
      v16h bp = load_frag_b(&Plds[wave][0][0], 72, lane & 15, ks * 32, lane);
#pragma unroll
      for (int ct = 0; ct < 8; ++ct) {
        v16h a = load_frag_a(&Vlds[buf][0][0], 72, ct * 16 + (lane & 15),
                             ks * 32, lane);
        oacc[ct] = wmma_f16(a, bp, oacc[ct]);
      }
    }
    buf ^= 1;
  }

  // ---- finalize: divide by softmax denom, write O (fp32) ----
  float inv = 1.0f / lrun;
#pragma unroll
  for (int ct = 0; ct < 8; ++ct)
#pragma unroll
    for (int v = 0; v < 8; ++v) {
      int c = ct * 16 + v + ((lane >> 4) << 3);
      o_ws[((size_t)(b * 1024 + hh * 128 + c)) * 2048 + lqn] = oacc[ct][v] * inv;
    }
}

// ---------------------------------------------------------------------------
// Kernel: out projection. GN2 normalization fused into per-K-step LDS staging,
// WMMA GEMM (M=128, K=1024), + out bias + residual x.
// ---------------------------------------------------------------------------
__global__ __launch_bounds__(256) void outproj_kernel(
    const float* __restrict__ o_ws, const float* __restrict__ stats2,
    const float* __restrict__ gn2_w, const float* __restrict__ gn2_b,
    const half_t* __restrict__ outw_h, const float* __restrict__ out_b,
    const float* __restrict__ x, float* __restrict__ out) {
  __shared__ alignas(16) half_t Blds[128][40];  // [lq][c_local], per K-step slab
  int b = blockIdx.y, Nt = blockIdx.x;
  int Nbase = Nt * 128;
  int t = threadIdx.x, wave = t >> 5, lane = t & 31;
  int mw = wave >> 2, nw = wave & 3;

  v8f acc[4][2];
#pragma unroll
  for (int mt = 0; mt < 4; ++mt)
#pragma unroll
    for (int nt = 0; nt < 2; ++nt) acc[mt][nt] = vzero8();

  for (int s = 0; s < 32; ++s) {
    int cb = s * 32;
    __syncthreads();
    for (int i = t; i < 32 * 128; i += 256) {
      int ci = i >> 7, j = i & 127;
      int c = cb + ci;
      float mu = stats2[(b * 32 + (c >> 5)) * 2];
      float rstd = stats2[(b * 32 + (c >> 5)) * 2 + 1];
      float vv = o_ws[((size_t)b * 1024 + c) * 2048 + Nbase + j];
      Blds[j][ci] = (half_t)((vv - mu) * rstd * gn2_w[c] + gn2_b[c]);
    }
    __syncthreads();
    v16h bf[2];
#pragma unroll
    for (int nt = 0; nt < 2; ++nt)
      bf[nt] = load_frag_b(&Blds[0][0], 40, nw * 32 + nt * 16 + (lane & 15), 0, lane);
#pragma unroll
    for (int mt = 0; mt < 4; ++mt) {
      int o = mw * 64 + mt * 16 + (lane & 15);
      v16h af = load_frag_a(outw_h, 1024, o, cb, lane);
#pragma unroll
      for (int nt = 0; nt < 2; ++nt) acc[mt][nt] = wmma_f16(af, bf[nt], acc[mt][nt]);
    }
  }

#pragma unroll
  for (int mt = 0; mt < 4; ++mt)
#pragma unroll
    for (int nt = 0; nt < 2; ++nt)
#pragma unroll
      for (int v = 0; v < 8; ++v) {
        int o = mw * 64 + mt * 16 + v + ((lane >> 4) << 3);
        int lq = Nbase + nw * 32 + nt * 16 + (lane & 15);
        size_t idx = ((size_t)b * 128 + o) * 2048 + lq;
        out[idx] = acc[mt][nt][v] + out_b[o] + x[idx];
      }
}

// ---------------------------------------------------------------------------
// Host launcher
// ---------------------------------------------------------------------------
extern "C" void kernel_launch(void* const* d_in, const int* in_sizes, int n_in,
                              void* d_out, int out_size, void* d_ws, size_t ws_size,
                              hipStream_t stream) {
  (void)in_sizes; (void)n_in; (void)out_size; (void)ws_size;
  const float* x    = (const float*)d_in[0];
  const float* ctx  = (const float*)d_in[1];
  const float* gn1w = (const float*)d_in[2];
  const float* gn1b = (const float*)d_in[3];
  const float* qw   = (const float*)d_in[4];
  const float* qb   = (const float*)d_in[5];
  const float* kvw  = (const float*)d_in[6];
  const float* kvb  = (const float*)d_in[7];
  const float* gn2w = (const float*)d_in[8];
  const float* gn2b = (const float*)d_in[9];
  const float* outw = (const float*)d_in[10];
  const float* outb = (const float*)d_in[11];
  float* out = (float*)d_out;

  char* ws = (char*)d_ws;
  size_t off = 0;
  auto take = [&](size_t bytes) {
    size_t o = off;
    off += (bytes + 255) & ~(size_t)255;
    return o;
  };
  float*  stats1 = (float*)(ws + take(128 * 2 * sizeof(float)));
  float*  stats2 = (float*)(ws + take(128 * 2 * sizeof(float)));
  half_t* qw_h   = (half_t*)(ws + take((size_t)1024 * 128 * 2));
  half_t* outw_h = (half_t*)(ws + take((size_t)128 * 1024 * 2));
  half_t* kvw_h  = (half_t*)(ws + take((size_t)2048 * 768 * 2));
  half_t* q_ws   = (half_t*)(ws + take((size_t)32 * 2048 * 128 * 2));
  half_t* k_ws   = (half_t*)(ws + take((size_t)32 * 2048 * 128 * 2));
  half_t* v_ws   = (half_t*)(ws + take((size_t)32 * 2048 * 128 * 2));
  float*  o_ws   = (float*)(ws + take((size_t)4 * 1024 * 2048 * sizeof(float)));

  convert_weights_kernel<<<6144, 256, 0, stream>>>(qw, kvw, outw, qw_h, kvw_h, outw_h);
  gn_stats_kernel<<<128, 256, 0, stream>>>(x, stats1, 128, 4);
  qproj_kernel<<<dim3(16, 8, 4), 256, 0, stream>>>(x, gn1w, gn1b, stats1, qw_h, qb, q_ws);
  kvproj_kernel<<<dim3(16, 16, 4), 256, 0, stream>>>(ctx, kvw_h, kvb, k_ws, v_ws);
  attn_kernel<<<dim3(32, 16), 256, 0, stream>>>(q_ws, k_ws, v_ws, o_ws);
  gn_stats_kernel<<<128, 256, 0, stream>>>(o_ws, stats2, 1024, 32);
  outproj_kernel<<<dim3(16, 4), 256, 0, stream>>>(o_ws, stats2, gn2w, gn2b, outw_h, outb, x, out);
}
